// MHSA_13511967113499
// MI455X (gfx1250) — matrix operand order repaired and checked by reading, not compiled, not run on hardware
//
#include <hip/hip_runtime.h>
#include <hip/hip_bf16.h>

typedef __attribute__((ext_vector_type(16))) __bf16 v16bf;
typedef __attribute__((ext_vector_type(8)))  __bf16 v8bf;
typedef __attribute__((ext_vector_type(8)))  float  v8f;

#define WMMA_BF16(a, b, c) \
    __builtin_amdgcn_wmma_f32_16x16x32_bf16(false, (a), false, (b), (short)0, (c), false, false)

#define DIM   512
#define HEADS 8
#define HD    64
#define BATCH 4
#define SEQ   2048
#define ROWS  (BATCH * SEQ)   // 8192

// ---- gfx1250 async copy-to-LDS path (guarded; falls back to plain stores) ----
#if __has_builtin(__builtin_amdgcn_global_load_async_to_lds_b128) && \
    __has_builtin(__builtin_amdgcn_s_wait_asynccnt)
#define USE_ASYNC_LDS 1
#else
#define USE_ASYNC_LDS 0
#endif

__device__ __forceinline__ void copy8bf_to_lds(const __bf16* g, __bf16* l) {
#if USE_ASYNC_LDS
    // builtin signature (from clang diagnostic): (v4i addrspace(1)*, v4i addrspace(3)*, imm, imm)
    typedef int i128 __attribute__((vector_size(16)));
    __builtin_amdgcn_global_load_async_to_lds_b128(
        (__attribute__((address_space(1))) i128*)g,
        (__attribute__((address_space(3))) i128*)l,
        0, 0);
#else
    *(v8bf*)l = *(const v8bf*)g;
#endif
}
__device__ __forceinline__ void async_copy_fence() {
#if USE_ASYNC_LDS
    __builtin_amdgcn_s_wait_asynccnt(0);
#endif
}

// ---------------------------------------------------------------------------
// Kernel 0: elementwise fp32 -> bf16 (one pass; removes all cvts from GEMM loops)
// ---------------------------------------------------------------------------
__global__ __launch_bounds__(256) void cvt_bf16_kernel(
    const float* __restrict__ src, __bf16* __restrict__ dst, int n8)
{
    const int i = blockIdx.x * 256 + threadIdx.x;
    if (i >= n8) return;
    const int base = i * 8;
    v8bf v;
#pragma unroll
    for (int j = 0; j < 8; ++j) v[j] = (__bf16)src[base + j];
    *(v8bf*)(dst + base) = v;
}

// ---------------------------------------------------------------------------
// Kernel 1: qkv = x @ Wqkv^T (all bf16), fused L2-norm of Q/K, V transposed.
// Weight tile (64 cols x 512 k) staged once per block into LDS (async path).
// Block = 8 waves; each wave: 16 rows x 64 cols; K-loop 16 x 32.
// ---------------------------------------------------------------------------
#define WPAD 8   // pad LDS rows so 16 lanes reading stride-row hit distinct banks

__global__ __launch_bounds__(256) void qkv_norm_kernel(
    const __bf16* __restrict__ x,      // [8192, 512]
    const __bf16* __restrict__ wqkv,   // [1536, 512]
    __bf16* __restrict__ Qbf,          // [4,8,2048,64]
    __bf16* __restrict__ Kbf,          // [4,8,2048,64]
    __bf16* __restrict__ Vt)           // [4,8,64,2048]
{
    __shared__ __bf16 wtile[64][DIM + WPAD];   // ~66.5 KB

    const int tid  = threadIdx.x;
    const int lane = tid & 31;
    const int wave = tid >> 5;
    const int hi   = lane >> 4;
    const int ln   = lane & 15;

    const int nCB     = (3 * DIM) / 64;     // 24
    const int cb      = blockIdx.x % nCB;
    const int rb      = blockIdx.x / nCB;
    const int rowBase = rb * 128 + wave * 16;
    const int colBase = cb * 64;

    // ---- stage 64x512 bf16 weight tile: 256 threads x 256B each ----
    {
        const int row = tid >> 2;           // 0..63
        const int seg = tid & 3;            // 0..3 (128 bf16 each)
        const __bf16* src = wqkv + (size_t)(colBase + row) * DIM + seg * 128;
        __bf16* dstl = &wtile[row][seg * 128];
#pragma unroll
        for (int c = 0; c < 16; ++c)
            copy8bf_to_lds(src + c * 8, dstl + c * 8);
    }
    async_copy_fence();
    __syncthreads();

    v8f acc[4] = {};
    for (int kk = 0; kk < DIM; kk += 32) {
        const __bf16* arow = x + (size_t)(rowBase + ln) * DIM + kk + 16 * hi;
        v16bf a;
#pragma unroll
        for (int j = 0; j < 8; ++j) { a[j] = arow[j]; a[8 + j] = arow[8 + j]; }
#pragma unroll
        for (int t = 0; t < 4; ++t) {
            const __bf16* brow = &wtile[t * 16 + ln][kk + 16 * hi];
            v16bf bf;
#pragma unroll
            for (int j = 0; j < 16; ++j) bf[j] = brow[j];
            acc[t] = WMMA_BF16(a, bf, acc[t]);
        }
    }

    const int tIdx = colBase >> 9;      // 0=Q 1=K 2=V
    const int h    = (colBase >> 6) & 7;

    if (tIdx < 2) {
        float inv[8];
#pragma unroll
        for (int r = 0; r < 8; ++r) {
            float s = 0.f;
#pragma unroll
            for (int t = 0; t < 4; ++t) { float v = acc[t][r]; s += v * v; }
#pragma unroll
            for (int m = 1; m < 16; m <<= 1) s += __shfl_xor(s, m, 16);
            inv[r] = 1.f / fmaxf(sqrtf(s), 1e-12f);
        }
        __bf16* dst = (tIdx == 0) ? Qbf : Kbf;
#pragma unroll
        for (int r = 0; r < 8; ++r) {
            const int m = rowBase + r + 8 * hi;
            const int b = m >> 11, n = m & (SEQ - 1);
            const size_t base = (((size_t)(b * HEADS + h) * SEQ) + n) * HD;
#pragma unroll
            for (int t = 0; t < 4; ++t)
                dst[base + t * 16 + ln] = (__bf16)(acc[t][r] * inv[r]);
        }
    } else {
#pragma unroll
        for (int r = 0; r < 8; ++r) {
            const int m = rowBase + r + 8 * hi;
            const int b = m >> 11, n = m & (SEQ - 1);
#pragma unroll
            for (int t = 0; t < 4; ++t) {
                const int d = t * 16 + ln;
                Vt[(((size_t)(b * HEADS + h) * HD) + d) * SEQ + n] = (__bf16)acc[t][r];
            }
        }
    }
}

// ---------------------------------------------------------------------------
// Kernel 2: flash attention per (b,h). K/V 32-key tiles staged per block in
// LDS (async path, shared by all 8 waves). Each wave: 16 queries x d=64 out.
// ---------------------------------------------------------------------------
__global__ __launch_bounds__(256) void attn_kernel(
    const __bf16* __restrict__ Q,     // [4,8,2048,64] normalized
    const __bf16* __restrict__ K,     // [4,8,2048,64] normalized
    const __bf16* __restrict__ Vt,    // [4,8,64,2048]
    const float*  __restrict__ temp,  // [8]
    __bf16* __restrict__ AO)          // [4,2048,512]
{
    __shared__ __bf16 Kst[32][HD + 8];   // 32 keys x 64 dims, padded rows
    __shared__ __bf16 Vst[HD][32 + 8];   // 64 dims x 32 keys, padded rows
    __shared__ __bf16 plds[8][16][40];   // per-wave P transpose buffer

    const int tid  = threadIdx.x;
    const int lane = tid & 31;
    const int wave = tid >> 5;
    const int hi   = lane >> 4;
    const int ln   = lane & 15;

    const int bh    = blockIdx.x >> 4;
    const int qblk  = blockIdx.x & 15;
    const int b     = bh >> 3;
    const int h     = bh & 7;
    const int qBase = qblk * 128 + wave * 16;
    const float tscale = temp[h];

    const __bf16* Qh = Q  + (size_t)bh * SEQ * HD;
    const __bf16* Kh = K  + (size_t)bh * SEQ * HD;
    const __bf16* Vh = Vt + (size_t)bh * HD * SEQ;

    // Q A-fragments (held in registers for the whole key loop)
    v16bf qa[2];
#pragma unroll
    for (int s = 0; s < 2; ++s) {
        const __bf16* qrow = Qh + (size_t)(qBase + ln) * HD + s * 32 + 16 * hi;
#pragma unroll
        for (int j = 0; j < 8; ++j) { qa[s][j] = qrow[j]; qa[s][8 + j] = qrow[8 + j]; }
    }

    v8f o[4] = {};
    float mrun[8], lrun[8];
#pragma unroll
    for (int r = 0; r < 8; ++r) { mrun[r] = -1e30f; lrun[r] = 0.f; }

    __bf16 (*pw)[40] = plds[wave];

    for (int kb = 0; kb < SEQ; kb += 32) {
        // ---- cooperative stage of K tile (32x64) and V tile (64x32) ----
        {
            const int krow = tid >> 3, kch = tid & 7;     // 32 rows x 8 chunks
            copy8bf_to_lds(Kh + (size_t)(kb + krow) * HD + kch * 8,
                           &Kst[krow][kch * 8]);
            const int vrow = tid >> 2, vch = tid & 3;     // 64 rows x 4 chunks
            copy8bf_to_lds(Vh + (size_t)vrow * SEQ + kb + vch * 8,
                           &Vst[vrow][vch * 8]);
        }
        async_copy_fence();
        __syncthreads();

        // ---- S = Q @ K^T for two 16-key tiles ----
        v8f c0 = {}, c1 = {};
#pragma unroll
        for (int s = 0; s < 2; ++s) {
            const __bf16* k0 = &Kst[ln][s * 32 + 16 * hi];
            v16bf b0;
#pragma unroll
            for (int j = 0; j < 16; ++j) b0[j] = k0[j];
            c0 = WMMA_BF16(qa[s], b0, c0);

            const __bf16* k1 = &Kst[16 + ln][s * 32 + 16 * hi];
            v16bf b1;
#pragma unroll
            for (int j = 0; j < 16; ++j) b1[j] = k1[j];
            c1 = WMMA_BF16(qa[s], b1, c1);
        }

        // ---- online softmax (rows = r + 8*hi, reduce across 16 lanes) ----
        float corr[8];
#pragma unroll
        for (int r = 0; r < 8; ++r) {
            float s0 = c0[r] * tscale, s1 = c1[r] * tscale;
            float mt = fmaxf(s0, s1);
#pragma unroll
            for (int msk = 1; msk < 16; msk <<= 1) mt = fmaxf(mt, __shfl_xor(mt, msk, 16));
            const float mnew = fmaxf(mrun[r], mt);
            const float cr   = __expf(mrun[r] - mnew);
            const float p0   = __expf(s0 - mnew);
            const float p1   = __expf(s1 - mnew);
            c0[r] = p0; c1[r] = p1;
            float rs = p0 + p1;
#pragma unroll
            for (int msk = 1; msk < 16; msk <<= 1) rs += __shfl_xor(rs, msk, 16);
            lrun[r] = lrun[r] * cr + rs;
            mrun[r] = mnew;
            corr[r] = cr;
        }
#pragma unroll
        for (int t = 0; t < 4; ++t)
#pragma unroll
            for (int r = 0; r < 8; ++r) o[t][r] *= corr[r];

        // ---- transpose P through wave-local LDS (C-layout -> A-layout) ----
#pragma unroll
        for (int r = 0; r < 8; ++r) {
            pw[r + 8 * hi][ln]      = (__bf16)c0[r];
            pw[r + 8 * hi][16 + ln] = (__bf16)c1[r];
        }
        v16bf pa;
#pragma unroll
        for (int j = 0; j < 8; ++j) {
            pa[j]     = pw[ln][16 * hi + j];
            pa[8 + j] = pw[ln][16 * hi + 8 + j];
        }

        // ---- O += P @ V from staged V^T tile ----
#pragma unroll
        for (int t = 0; t < 4; ++t) {
            const __bf16* vrow = &Vst[t * 16 + ln][16 * hi];
            v16bf vb;
#pragma unroll
            for (int j = 0; j < 16; ++j) vb[j] = vrow[j];
            o[t] = WMMA_BF16(pa, vb, o[t]);
        }
        __syncthreads();   // protect Kst/Vst before next-iteration staging
    }

#pragma unroll
    for (int r = 0; r < 8; ++r) {
        const float inv = 1.f / lrun[r];
        const int n = qBase + r + 8 * hi;
        const size_t base = ((size_t)b * SEQ + n) * DIM + h * HD;
#pragma unroll
        for (int t = 0; t < 4; ++t)
            AO[base + t * 16 + ln] = (__bf16)(o[t][r] * inv);
    }
}

// ---------------------------------------------------------------------------
// Kernel 3: out = AO @ Wout^T, weight tile staged in LDS, fp32 output.
// ---------------------------------------------------------------------------
__global__ __launch_bounds__(256) void proj_kernel(
    const __bf16* __restrict__ AO,    // [8192, 512] bf16
    const __bf16* __restrict__ wout,  // [512, 512] bf16
    float* __restrict__ out)          // [8192, 512]
{
    __shared__ __bf16 wtile[64][DIM + WPAD];

    const int tid  = threadIdx.x;
    const int lane = tid & 31;
    const int wave = tid >> 5;
    const int hi   = lane >> 4;
    const int ln   = lane & 15;

    const int nCB     = DIM / 64;     // 8
    const int cb      = blockIdx.x % nCB;
    const int rb      = blockIdx.x / nCB;
    const int rowBase = rb * 128 + wave * 16;
    const int colBase = cb * 64;

    {
        const int row = tid >> 2;
        const int seg = tid & 3;
        const __bf16* src = wout + (size_t)(colBase + row) * DIM + seg * 128;
        __bf16* dstl = &wtile[row][seg * 128];
#pragma unroll
        for (int c = 0; c < 16; ++c)
            copy8bf_to_lds(src + c * 8, dstl + c * 8);
    }
    async_copy_fence();
    __syncthreads();

    v8f acc[4] = {};
    for (int kk = 0; kk < DIM; kk += 32) {
        const __bf16* arow = AO + (size_t)(rowBase + ln) * DIM + kk + 16 * hi;
        v16bf a;
#pragma unroll
        for (int j = 0; j < 8; ++j) { a[j] = arow[j]; a[8 + j] = arow[8 + j]; }
#pragma unroll
        for (int t = 0; t < 4; ++t) {
            const __bf16* brow = &wtile[t * 16 + ln][kk + 16 * hi];
            v16bf bf;
#pragma unroll
            for (int j = 0; j < 16; ++j) bf[j] = brow[j];
            acc[t] = WMMA_BF16(a, bf, acc[t]);
        }
    }

#pragma unroll
    for (int r = 0; r < 8; ++r) {
        const int m = rowBase + r + 8 * hi;
#pragma unroll
        for (int t = 0; t < 4; ++t)
            out[(size_t)m * DIM + colBase + t * 16 + ln] = acc[t][r];
    }
}

// ---------------------------------------------------------------------------
extern "C" void kernel_launch(void* const* d_in, const int* in_sizes, int n_in,
                              void* d_out, int out_size, void* d_ws, size_t ws_size,
                              hipStream_t stream) {
    const float* x    = (const float*)d_in[0];   // [4,2048,512]
    const float* wqkv = (const float*)d_in[1];   // [1536,512]
    const float* wout = (const float*)d_in[2];   // [512,512]
    const float* temp = (const float*)d_in[3];   // [8]
    float* out = (float*)d_out;

    const size_t nx  = (size_t)ROWS * DIM;          // 4,194,304
    const size_t nwq = (size_t)3 * DIM * DIM;       //   786,432
    const size_t nwo = (size_t)DIM * DIM;           //   262,144
    const size_t nqk = (size_t)BATCH * HEADS * SEQ * HD;  // 4,194,304

    char* ws = (char*)d_ws;
    size_t off = 0;
    __bf16* xbf    = (__bf16*)(ws + off); off += 2 * nx;
    __bf16* wqkvbf = (__bf16*)(ws + off); off += 2 * nwq;
    __bf16* woutbf = (__bf16*)(ws + off); off += 2 * nwo;
    __bf16* Qbf    = (__bf16*)(ws + off); off += 2 * nqk;
    __bf16* Kbf    = (__bf16*)(ws + off); off += 2 * nqk;
    __bf16* Vt     = (__bf16*)(ws + off); off += 2 * nqk;
    __bf16* AO     = (__bf16*)(ws + off); off += 2 * nqk;

    // one-shot fp32 -> bf16 conversion pre-pass (memory bound, ~1 us)
    cvt_bf16_kernel<<<(int)(nx  / 8 / 256), 256, 0, stream>>>(x,    xbf,    (int)(nx  / 8));
    cvt_bf16_kernel<<<(int)(nwq / 8 / 256), 256, 0, stream>>>(wqkv, wqkvbf, (int)(nwq / 8));
    cvt_bf16_kernel<<<(int)(nwo / 8 / 256), 256, 0, stream>>>(wout, woutbf, (int)(nwo / 8));

    qkv_norm_kernel<<<(ROWS / 128) * ((3 * DIM) / 64), 256, 0, stream>>>(
        xbf, wqkvbf, Qbf, Kbf, Vt);

    attn_kernel<<<BATCH * HEADS * (SEQ / 128), 256, 0, stream>>>(
        Qbf, Kbf, Vt, temp, AO);

    proj_kernel<<<(ROWS / 128) * (DIM / 64), 256, 0, stream>>>(AO, woutbf, out);
}
